// DeltaProductBlock_89232240542351
// MI455X (gfx1250) — matrix operand order repaired and checked by reading, not compiled, not run on hardware
//
#include <hip/hip_runtime.h>
#include <cstdint>

#define DEV __device__ __forceinline__

typedef __attribute__((ext_vector_type(16))) __bf16 v16bf;
typedef __attribute__((ext_vector_type(8)))  float  v8f;
typedef __attribute__((ext_vector_type(4)))  float  v4f;
typedef __attribute__((ext_vector_type(4)))  unsigned int v4u;

namespace {
constexpr int cB  = 2;
constexpr int cT  = 2048;
constexpr int cD  = 2048;
constexpr int cH  = 16;
constexpr float cSCALE = 0.08838834764831845f;  // 128^-0.5
}

union Frag { v16bf v; v4u q[2]; };

DEV unsigned short f2bf(float f) {
  unsigned int u = __float_as_uint(f);
  u += 0x7FFFu + ((u >> 16) & 1u);
  return (unsigned short)(u >> 16);
}

// ---------------------------------------------------------------------------
// CDNA5 async global->LDS copies (ASYNCcnt-tracked).
// Low 32 bits of a flat LDS address are the wave-relative LDS offset
// (shared aperture base has zero low bits).
// ---------------------------------------------------------------------------
DEV void async_b32(const void* g, void* l) {
  unsigned loff = (unsigned)(unsigned long long)l;
  unsigned long long ga = (unsigned long long)g;
  asm volatile("global_load_async_to_lds_b32 %0, %1, off"
               :: "v"(loff), "v"(ga) : "memory");
}
DEV void async_b128(const void* g, void* l) {
  unsigned loff = (unsigned)(unsigned long long)l;
  unsigned long long ga = (unsigned long long)g;
  asm volatile("global_load_async_to_lds_b128 %0, %1, off"
               :: "v"(loff), "v"(ga) : "memory");
}
DEV void wait_async0() {
  asm volatile("s_wait_asynccnt 0x0" ::: "memory");
}

// ---------------------------------------------------------------------------
// fp32 -> bf16 conversion (packed, 8 elems/thread)
// ---------------------------------------------------------------------------
__global__ void cvt_f32_bf16(const float* __restrict__ src,
                             unsigned short* __restrict__ dst, long long n) {
  long long i = ((long long)blockIdx.x * blockDim.x + threadIdx.x) * 8;
  if (i >= n) return;
  v4f a = *(const v4f*)(src + i);
  v4f b = *(const v4f*)(src + i + 4);
  v4u o;
  o[0] = (unsigned)f2bf(a[0]) | ((unsigned)f2bf(a[1]) << 16);
  o[1] = (unsigned)f2bf(a[2]) | ((unsigned)f2bf(a[3]) << 16);
  o[2] = (unsigned)f2bf(b[0]) | ((unsigned)f2bf(b[1]) << 16);
  o[3] = (unsigned)f2bf(b[2]) | ((unsigned)f2bf(b[3]) << 16);
  *(v4u*)(dst + i) = o;
}

// ---------------------------------------------------------------------------
// bf16 WMMA GEMM: C[M,N] (f32) = A[M,K] * B[K,N], all row-major, dims % 128==0
// 128x128x64 block tile, 8 waves (4x2), double-buffered LDS (2x32KB).
// A tile staged with async global->LDS b128; B tile transposed with packed
// ds_store_b32 (two adjacent k-rows per dword). 16 WMMAs per barrier.
// ---------------------------------------------------------------------------
DEV void stage_A(const unsigned short* __restrict__ A,
                 unsigned short (*la)[64], size_t bm, int K, int k0,
                 int arow, int ac32) {
  const unsigned short* g0 = &A[(bm + arow) * (size_t)K + k0 + ac32];
#pragma unroll
  for (int j = 0; j < 4; ++j)
    async_b128(g0 + 8 * j, &la[arow][ac32 + 8 * j]);
}

DEV void stage_B(const unsigned short* __restrict__ Bm,
                 unsigned short (*lb)[64], size_t bn, int N, int k0, int tid) {
#pragma unroll
  for (int rep = 0; rep < 2; ++rep) {
    const int task = tid + rep * 256;       // 512 tasks: 32 k-pairs x 16 n-chunks
    const int k2 = (task >> 4) * 2;         // even k row
    const int n0 = (task & 15) * 8;         // 8-column chunk
    v4u r0 = *(const v4u*)&Bm[(size_t)(k0 + k2) * (size_t)N + bn + n0];
    v4u r1 = *(const v4u*)&Bm[(size_t)(k0 + k2 + 1) * (size_t)N + bn + n0];
#pragma unroll
    for (int j = 0; j < 4; ++j) {
      const unsigned lo0 = r0[j] & 0xFFFFu, hi0 = r0[j] >> 16;
      const unsigned lo1 = r1[j] & 0xFFFFu, hi1 = r1[j] >> 16;
      *(unsigned*)&lb[n0 + 2 * j    ][k2] = lo0 | (lo1 << 16);
      *(unsigned*)&lb[n0 + 2 * j + 1][k2] = hi0 | (hi1 << 16);
    }
  }
}

__global__ __launch_bounds__(256) void gemm_bf16(
    const unsigned short* __restrict__ A, const unsigned short* __restrict__ Bm,
    float* __restrict__ C, int M, int N, int K) {
  __shared__ unsigned short lA[2][128][64];  // [buf][m][k]   32KB
  __shared__ unsigned short lB[2][128][64];  // [buf][n][k]   32KB (transposed)
  const int tid  = threadIdx.x;
  const int lane = tid & 31;
  const int wave = tid >> 5;
  const int wm   = wave & 3;   // 4 wave rows  -> 32 M-rows each
  const int wn   = wave >> 2;  // 2 wave cols  -> 64 N-cols each
  const size_t bm = (size_t)blockIdx.y * 128;
  const size_t bn = (size_t)blockIdx.x * 128;

  const int arow = tid >> 1;
  const int ac32 = (tid & 1) * 32;

  const v8f vzero = {0.f, 0.f, 0.f, 0.f, 0.f, 0.f, 0.f, 0.f};
  v8f acc[2][4];
#pragma unroll
  for (int i = 0; i < 2; ++i)
#pragma unroll
    for (int j = 0; j < 4; ++j) acc[i][j] = vzero;

  const int nkt = K >> 6;  // 64-wide K steps
  stage_A(A, lA[0], bm, K, 0, arow, ac32);
  stage_B(Bm, lB[0], bn, N, 0, tid);
  wait_async0();
  __syncthreads();

  const int mlane = lane & 15;
  const int akb = (lane < 16) ? 0 : 8;   // ISA A-frag K-chunk base
  const int bkb = (lane < 16) ? 0 : 8;   // ISA B-frag K-chunk base

  for (int kt = 0; kt < nkt; ++kt) {
    const int cur = kt & 1;
    if (kt + 1 < nkt) {
      stage_A(A, lA[cur ^ 1], bm, K, (kt + 1) << 6, arow, ac32);
      stage_B(Bm, lB[cur ^ 1], bn, N, (kt + 1) << 6, tid);
    }
#pragma unroll
    for (int kk = 0; kk < 64; kk += 32) {
      Frag af[2], bfr[4];
#pragma unroll
      for (int i = 0; i < 2; ++i) {
        const int r = wm * 32 + i * 16 + mlane;
        af[i].q[0] = *(const v4u*)&lA[cur][r][kk + akb];
        af[i].q[1] = *(const v4u*)&lA[cur][r][kk + akb + 16];
      }
#pragma unroll
      for (int j = 0; j < 4; ++j) {
        const int c = wn * 64 + j * 16 + mlane;
        bfr[j].q[0] = *(const v4u*)&lB[cur][c][kk + bkb];
        bfr[j].q[1] = *(const v4u*)&lB[cur][c][kk + bkb + 16];
      }
#pragma unroll
      for (int i = 0; i < 2; ++i)
#pragma unroll
        for (int j = 0; j < 4; ++j)
          acc[i][j] = __builtin_amdgcn_wmma_f32_16x16x32_bf16(
              false, af[i].v, false, bfr[j].v, (short)0, acc[i][j], false, false);
    }
    wait_async0();
    __syncthreads();
  }

  const int rofs = (lane < 16) ? 0 : 8;  // C/D layout: VGPR r -> row r / r+8
#pragma unroll
  for (int i = 0; i < 2; ++i)
#pragma unroll
    for (int j = 0; j < 4; ++j)
#pragma unroll
      for (int r = 0; r < 8; ++r) {
        const size_t row = bm + wm * 32 + i * 16 + r + rofs;
        const size_t col = bn + wn * 64 + j * 16 + mlane;
        C[row * (size_t)N + col] = acc[i][j][r];
      }
}

// ---------------------------------------------------------------------------
// beta = 2*sigmoid(x@Wb), g = -exp(A_log)*softplus(x@Wa + dt_bias)
// one block per (b,t) row
// ---------------------------------------------------------------------------
__global__ __launch_bounds__(256) void betag_kernel(
    const float* __restrict__ x, const float* __restrict__ Wb,
    const float* __restrict__ Wa, const float* __restrict__ A_log,
    const float* __restrict__ dt_bias, float* __restrict__ betaB,
    float* __restrict__ gB) {
  __shared__ float xr[2048];
  __shared__ float red[256];
  const long long bt = blockIdx.x;
  const int tid = threadIdx.x;
  for (int i = tid; i < 2048; i += 256) xr[i] = x[bt * 2048 + i];
  __syncthreads();
  const int col = tid >> 2;   // 0..63 (48 beta + 16 g)
  const int part = tid & 3;
  float acc = 0.f;
  if (col < 48) {
    for (int kk = part * 512; kk < part * 512 + 512; ++kk)
      acc = fmaf(xr[kk], Wb[(size_t)kk * 48 + col], acc);
  } else {
    const int h = col - 48;
    for (int kk = part * 512; kk < part * 512 + 512; ++kk)
      acc = fmaf(xr[kk], Wa[(size_t)kk * 16 + h], acc);
  }
  red[tid] = acc;
  __syncthreads();
  if (part == 0) {
    const float s = red[tid] + red[tid + 1] + red[tid + 2] + red[tid + 3];
    if (col < 48) {
      betaB[bt * 48 + col] = 2.f / (1.f + __expf(-s));
    } else {
      const int h = col - 48;
      const float z = s + dt_bias[h];
      const float sp = (z > 20.f) ? z : log1pf(__expf(z));
      gB[bt * 16 + h] = -__expf(A_log[h]) * sp;
    }
  }
}

// ---------------------------------------------------------------------------
// causal depthwise conv(K=4) + SiLU + optional L2-norm over 128-channel groups
// one wave (32 lanes * 4 ch) per [b,t,group] vector
// ---------------------------------------------------------------------------
__global__ __launch_bounds__(256) void conv_silu_norm(
    const float* __restrict__ in, const float* __restrict__ w,
    float* __restrict__ out, int C, int do_norm, float scale) {
  const int lane = threadIdx.x & 31;
  const int wave = threadIdx.x >> 5;
  const int gpr = C >> 7;
  const long long vec = (long long)blockIdx.x * 8 + wave;
  const int grp = (int)(vec % gpr);
  const long long bt = vec / gpr;
  const int t = (int)(bt % cT);
  const long long rowbase = bt - t;
  float y[4];
#pragma unroll
  for (int j = 0; j < 4; ++j) {
    const int c = grp * 128 + lane + j * 32;
    float acc = 0.f;
#pragma unroll
    for (int tap = 0; tap < 4; ++tap) {
      const int tt = t - 3 + tap;
      const float xv = (tt >= 0) ? in[(rowbase + tt) * (long long)C + c] : 0.f;
      acc = fmaf(xv, w[c * 4 + tap], acc);
    }
    y[j] = acc / (1.f + __expf(-acc));  // SiLU
  }
  float mult = scale;
  if (do_norm) {
    float ss = y[0] * y[0] + y[1] * y[1] + y[2] * y[2] + y[3] * y[3];
#pragma unroll
    for (int m = 16; m >= 1; m >>= 1) ss += __shfl_xor(ss, m, 32);
    mult = rsqrtf(ss + 1e-6f) * scale;
  }
#pragma unroll
  for (int j = 0; j < 4; ++j) {
    const int c = grp * 128 + lane + j * 32;
    out[bt * (long long)C + c] = y[j] * mult;
  }
}

// ---------------------------------------------------------------------------
// gated delta-product recurrence. One block (128 thr) per (b,h); lane owns one
// dv column: S[dk][dv] lives as 128 fp32 in VGPRs, dk-contraction is in-lane.
// Per-step operands double-buffered in LDS via async global->LDS copies.
// ---------------------------------------------------------------------------
struct ScanStage {
  float q[128];
  float k[3][128];
  float v[3][128];
  float b[4];  // b[0..2] = beta, b[3] = g
};

DEV void scan_stage(ScanStage* s, const float* __restrict__ q,
                    const float* __restrict__ k, const float* __restrict__ v,
                    const float* __restrict__ beta, const float* __restrict__ g,
                    long long bt, int h, int dv) {
  async_b32(&q[(bt * 16 + h) * 128 + dv], &s->q[dv]);
#pragma unroll
  for (int i = 0; i < 3; ++i) {
    async_b32(&k[(bt * 48 + i * 16 + h) * 128 + dv], &s->k[i][dv]);
    async_b32(&v[(bt * 48 + i * 16 + h) * 128 + dv], &s->v[i][dv]);
  }
  if (dv < 3) async_b32(&beta[bt * 48 + dv * 16 + h], &s->b[dv]);
  if (dv == 3) async_b32(&g[bt * 16 + h], &s->b[3]);
}

__global__ __launch_bounds__(128) void scan_kernel(
    const float* __restrict__ q, const float* __restrict__ k,
    const float* __restrict__ v, const float* __restrict__ beta,
    const float* __restrict__ g, float* __restrict__ o) {
  __shared__ ScanStage st[2];
  const int b = blockIdx.x >> 4;
  const int h = blockIdx.x & 15;
  const int dv = threadIdx.x;
  const long long bt0 = (long long)b * cT;
  float S[128];
#pragma unroll
  for (int dk = 0; dk < 128; ++dk) S[dk] = 0.f;

  scan_stage(&st[0], q, k, v, beta, g, bt0, h, dv);
  wait_async0();
  __syncthreads();

  for (int t = 0; t < cT; ++t) {
    const int cur = t & 1;
    if (t + 1 < cT)
      scan_stage(&st[cur ^ 1], q, k, v, beta, g, bt0 + t + 1, h, dv);
    if (t + 2 < cT) {  // keep t+2 lines warm in L2
      const long long btn = bt0 + t + 2;
      __builtin_prefetch(&q[(btn * 16 + h) * 128 + dv], 0, 0);
#pragma unroll
      for (int i = 0; i < 3; ++i) {
        __builtin_prefetch(&k[(btn * 48 + i * 16 + h) * 128 + dv], 0, 0);
        __builtin_prefetch(&v[(btn * 48 + i * 16 + h) * 128 + dv], 0, 0);
      }
    }
    const ScanStage* sc = &st[cur];

    const float dec = __expf(sc->b[3]);
#pragma unroll
    for (int dk = 0; dk < 128; ++dk) S[dk] *= dec;

#pragma unroll
    for (int i = 0; i < 3; ++i) {
      float ks = 0.f;
#pragma unroll
      for (int dk = 0; dk < 128; ++dk) ks = fmaf(sc->k[i][dk], S[dk], ks);
      const float u = sc->b[i] * (sc->v[i][dv] - ks);
#pragma unroll
      for (int dk = 0; dk < 128; ++dk) S[dk] = fmaf(sc->k[i][dk], u, S[dk]);
    }

    float ot = 0.f;
#pragma unroll
    for (int dk = 0; dk < 128; ++dk) ot = fmaf(sc->q[dk], S[dk], ot);
    o[((bt0 + t) * 16 + h) * 128 + dv] = ot;

    wait_async0();
    __syncthreads();
  }
}

// ---------------------------------------------------------------------------
// o <- o * rsqrt(mean(o^2)+eps) * norm_w * silu(gate), emit bf16 for final GEMM
// ---------------------------------------------------------------------------
__global__ __launch_bounds__(256) void onorm_kernel(
    const float* __restrict__ o, const float* __restrict__ gate,
    const float* __restrict__ normw, unsigned short* __restrict__ obf) {
  const int lane = threadIdx.x & 31;
  const int wave = threadIdx.x >> 5;
  const long long vec = (long long)blockIdx.x * 8 + wave;  // = bt*16 + h
  float y[4];
  float ss = 0.f;
#pragma unroll
  for (int j = 0; j < 4; ++j) {
    const int dvi = lane + j * 32;
    y[j] = o[vec * 128 + dvi];
    ss = fmaf(y[j], y[j], ss);
  }
#pragma unroll
  for (int m = 16; m >= 1; m >>= 1) ss += __shfl_xor(ss, m, 32);
  const float r = rsqrtf(ss * (1.f / 128.f) + 1e-5f);
#pragma unroll
  for (int j = 0; j < 4; ++j) {
    const int dvi = lane + j * 32;
    const float gt = gate[vec * 128 + dvi];
    const float sg = gt / (1.f + __expf(-gt));
    obf[vec * 128 + dvi] = f2bf(y[j] * r * normw[dvi] * sg);
  }
}

// ---------------------------------------------------------------------------
extern "C" void kernel_launch(void* const* d_in, const int* in_sizes, int n_in,
                              void* d_out, int out_size, void* d_ws, size_t ws_size,
                              hipStream_t stream) {
  (void)in_sizes; (void)n_in; (void)out_size; (void)ws_size;
  const float* x     = (const float*)d_in[0];
  const float* Wq    = (const float*)d_in[3];
  const float* Wk    = (const float*)d_in[4];
  const float* Wv    = (const float*)d_in[5];
  const float* Wb    = (const float*)d_in[6];
  const float* Wa    = (const float*)d_in[7];
  const float* A_log = (const float*)d_in[8];
  const float* dt_b  = (const float*)d_in[9];
  const float* convq = (const float*)d_in[10];
  const float* convk = (const float*)d_in[11];
  const float* convv = (const float*)d_in[12];
  const float* Wg    = (const float*)d_in[13];
  const float* normw = (const float*)d_in[14];
  const float* Wo    = (const float*)d_in[15];
  float* out = (float*)d_out;

  char* p = (char*)d_ws;
  auto alloc = [&](size_t bytes) -> void* {
    void* r = (void*)p;
    p += (bytes + 255) & ~(size_t)255;
    return r;
  };
  const size_t M = (size_t)cB * cT;  // 4096 rows
  unsigned short* xbf  = (unsigned short*)alloc(M * cD * 2);
  unsigned short* wqbf = (unsigned short*)alloc((size_t)cD * 2048 * 2);
  unsigned short* wkbf = (unsigned short*)alloc((size_t)cD * 6144 * 2);
  unsigned short* wvbf = (unsigned short*)alloc((size_t)cD * 6144 * 2);
  unsigned short* wgbf = (unsigned short*)alloc((size_t)cD * 2048 * 2);
  unsigned short* wobf = (unsigned short*)alloc((size_t)2048 * cD * 2);
  float* qraw  = (float*)alloc(M * 2048 * 4);
  float* kraw  = (float*)alloc(M * 6144 * 4);
  float* vraw  = (float*)alloc(M * 6144 * 4);
  float* qc    = (float*)alloc(M * 2048 * 4);
  float* kc    = (float*)alloc(M * 6144 * 4);
  float* vc    = (float*)alloc(M * 6144 * 4);
  float* gateB = (float*)alloc(M * 2048 * 4);
  float* betaB = (float*)alloc(M * 48 * 4);
  float* gB    = (float*)alloc(M * 16 * 4);
  float* oB    = (float*)alloc(M * 2048 * 4);
  unsigned short* obf = (unsigned short*)alloc(M * 2048 * 2);

  auto cvt = [&](const float* s, unsigned short* d, size_t n) {
    dim3 grid((unsigned)((n / 8 + 255) / 256));
    cvt_f32_bf16<<<grid, dim3(256), 0, stream>>>(s, d, (long long)n);
  };
  cvt(x, xbf, M * cD);
  cvt(Wq, wqbf, (size_t)cD * 2048);
  cvt(Wk, wkbf, (size_t)cD * 6144);
  cvt(Wv, wvbf, (size_t)cD * 6144);
  cvt(Wg, wgbf, (size_t)cD * 2048);
  cvt(Wo, wobf, (size_t)2048 * cD);

  auto gemm = [&](const unsigned short* A, const unsigned short* Bm, float* C,
                  int Mm, int Nn, int Kk) {
    dim3 grid((unsigned)(Nn / 128), (unsigned)(Mm / 128));
    gemm_bf16<<<grid, dim3(256), 0, stream>>>(A, Bm, C, Mm, Nn, Kk);
  };
  gemm(xbf, wqbf, qraw, 4096, 2048, 2048);
  gemm(xbf, wkbf, kraw, 4096, 6144, 2048);
  gemm(xbf, wvbf, vraw, 4096, 6144, 2048);
  gemm(xbf, wgbf, gateB, 4096, 2048, 2048);

  betag_kernel<<<dim3(4096), dim3(256), 0, stream>>>(x, Wb, Wa, A_log, dt_b,
                                                     betaB, gB);

  conv_silu_norm<<<dim3(8192),  dim3(256), 0, stream>>>(qraw, convq, qc, 2048, 1, cSCALE);
  conv_silu_norm<<<dim3(24576), dim3(256), 0, stream>>>(kraw, convk, kc, 6144, 1, 1.0f);
  conv_silu_norm<<<dim3(24576), dim3(256), 0, stream>>>(vraw, convv, vc, 6144, 0, 1.0f);

  scan_kernel<<<dim3(cB * cH), dim3(128), 0, stream>>>(qc, kc, vc, betaB, gB, oB);

  onorm_kernel<<<dim3(8192), dim3(256), 0, stream>>>(oB, gateB, normw, obf);

  gemm(obf, wobf, out, 4096, 2048, 2048);
}
